// HybridGCNGNN_52269751993057
// MI455X (gfx1250) — matrix operand order repaired and checked by reading, not compiled, not run on hardware
//
#include <hip/hip_runtime.h>

#define NEG_SLOPE 0.2f
#define BN_EPS 1e-5f

// ---------------------------------------------------------------------------
// CDNA5 WMMA types
// ---------------------------------------------------------------------------
typedef __attribute__((ext_vector_type(16))) __bf16         v16bf;
typedef __attribute__((ext_vector_type(8)))  float          v8f;
typedef __attribute__((ext_vector_type(4)))  unsigned int   v4u;
typedef __attribute__((ext_vector_type(2)))  unsigned int   v2u;

__device__ __forceinline__ unsigned short f2bf(float f) {
    union { __bf16 b; unsigned short u; } cv;
    cv.b = (__bf16)f;           // RNE float->bf16
    return cv.u;
}
__device__ __forceinline__ __bf16 bf_from_us(unsigned short u) {
    union { unsigned short u; __bf16 b; } cv;
    cv.u = u;
    return cv.b;
}

__device__ __forceinline__ void atomicMaxF(float* addr, float val) {
    // correct for mixed-sign updates when addr is initialized to -inf
    if (!(val < 0.f)) atomicMax((int*)addr, __float_as_int(val));
    else              atomicMin((unsigned int*)addr, __float_as_uint(val));
}

// ---------------------------------------------------------------------------
// f32 -> bf16 bulk convert (pairs; packed u32 stores)
// ---------------------------------------------------------------------------
__global__ void f32_to_bf16(const float* __restrict__ X, unsigned short* __restrict__ Y,
                            size_t npairs) {
    size_t i = (size_t)blockIdx.x * blockDim.x + threadIdx.x;
    if (i >= npairs) return;
    float a = X[2 * i], b = X[2 * i + 1];
    unsigned int p = (unsigned int)f2bf(a) | ((unsigned int)f2bf(b) << 16);
    ((unsigned int*)Y)[i] = p;
}

// ---------------------------------------------------------------------------
// Weight pre-pack: W[K,Nact] f32 (row-major)  ->  Bt[Npad,K] bf16 (transposed)
// ---------------------------------------------------------------------------
__global__ void pack_weight_bf16(const float* __restrict__ W,
                                 unsigned short* __restrict__ Bt,
                                 int K, int Nact, int Npad) {
    int idx = blockIdx.x * blockDim.x + threadIdx.x;
    if (idx >= Npad * K) return;
    int n = idx / K, k = idx - n * K;
    float v = (n < Nact) ? W[(size_t)k * Nact + n] : 0.f;
    Bt[idx] = f2bf(v);
}

// ---------------------------------------------------------------------------
// WMMA GEMM: C[M,Npad] = act( A[M,K] @ W + bias ), bf16 inputs, f32 accum.
// block = 256 thr = 8 waves; wave w -> 16x16 tile (blockIdx.y, blockIdx.x*8+w)
// Requirements: M % 16 == 0, K % 64 == 0.  act: 0=none, 1=relu
// ---------------------------------------------------------------------------
#define TILE_K 64
__global__ void __launch_bounds__(256)
wmma_gemm_bf16(const unsigned short* __restrict__ A,   // bf16 [M,K]
               const unsigned short* __restrict__ Bt,  // bf16 [Npad,K]
               const float* __restrict__ bias, float* __restrict__ C,
               int M, int K, int Npad, int Nact, int act) {
    __shared__ unsigned short lAs[16 * TILE_K];        // 2 KB

    const int tid  = threadIdx.x;
    const int wave = tid >> 5;
    const int lane = tid & 31;
    const int tm   = blockIdx.y;
    const int tn   = blockIdx.x * 8 + wave;
    const bool active = (tn * 16) < Npad;
    const int rowBase = tm * 16;
    const int nl = lane & 15;     // row (A) / col (B,C) within tile
    const int hs = lane >> 4;     // K-half select (A,B) / row-half (C)

    // staging: thread copies 4 consecutive bf16 (8B vector), fixed (row,col)
    const int si = tid * 4;
    const int sr = si >> 6;       // 0..15
    const int sc = si & 63;
    const unsigned short* aptr = A + (size_t)(rowBase + sr) * K + sc;
    const unsigned short* bptr =
        Bt + (size_t)((active ? tn : 0) * 16 + nl) * K + hs * 16;

    v8f acc = {};

    for (int k0 = 0; k0 < K; k0 += TILE_K) {
        __syncthreads();
        *(v2u*)&lAs[si] = *(const v2u*)(aptr + k0);
        __syncthreads();
        if (active) {
#pragma unroll
            for (int s = 0; s < 2; ++s) {
                // A operand: lane nl = row; VGPR v -> K = s*32+hs*16+v*2
                v16bf va;
#pragma unroll
                for (int v = 0; v < 8; ++v) {
                    int kk = (s << 5) + (hs << 4) + (v << 1);
                    va[2 * v]     = bf_from_us(lAs[(nl << 6) + kk]);
                    va[2 * v + 1] = bf_from_us(lAs[(nl << 6) + kk + 1]);
                }
                // B operand: 16 contiguous K (one aligned 32B load)
                union { v4u q[2]; v16bf b; } bl;
                const v4u* bp = (const v4u*)(bptr + k0 + (s << 5));
                bl.q[0] = bp[0];
                bl.q[1] = bp[1];
                acc = __builtin_amdgcn_wmma_f32_16x16x32_bf16(
                    false, va, false, bl.b, (short)0, acc, false, false);
            }
        }
    }

    if (active) {
#pragma unroll
        for (int r = 0; r < 8; ++r) {
            int m = rowBase + r + hs * 8;   // C layout: VGPR r -> row r (+8 hi lanes)
            int n = tn * 16 + nl;
            if (n < Nact) {
                float v = acc[r];
                if (bias) v += bias[n];
                if (act == 1) v = v > 0.f ? v : 0.f;
                C[(size_t)m * Npad + n] = v;
            }
        }
    }
}

// ---------------------------------------------------------------------------
// Degree / normalization
// ---------------------------------------------------------------------------
__global__ void deg_init(float* deg, int n) {
    int i = blockIdx.x * blockDim.x + threadIdx.x;
    if (i < n) deg[i] = 1.f;                       // self-loop
}
__global__ void deg_accum(const long long* __restrict__ dst, float* deg, int E) {
    int e = blockIdx.x * blockDim.x + threadIdx.x;
    if (e < E) atomicAdd(&deg[dst[e]], 1.f);
}
__global__ void deg_finalize(float* deg, int n) {  // deg -> rsqrt(deg) in place
    int i = blockIdx.x * blockDim.x + threadIdx.x;
    if (i < n) deg[i] = rsqrtf(deg[i]);
}

// ---------------------------------------------------------------------------
// GCN aggregation
// ---------------------------------------------------------------------------
__global__ void gcn_self(const float* __restrict__ h, const float* __restrict__ dinv,
                         float* __restrict__ out, int n, int F) {
    size_t idx = (size_t)blockIdx.x * blockDim.x + threadIdx.x;
    if (idx >= (size_t)n * F) return;
    int row = (int)(idx / F);
    float di = dinv[row];
    out[idx] = h[idx] * di * di;
}
__global__ void gcn_edge_agg(const float* __restrict__ h, const long long* __restrict__ ei,
                             const float* __restrict__ dinv, float* __restrict__ out,
                             int E, int F) {
    int e = blockIdx.x;
    long long s = ei[e], d = ei[(size_t)E + e];
    float c = dinv[s] * dinv[d];
    const float* hs = h + (size_t)s * F;
    float* od = out + (size_t)d * F;
    for (int f = threadIdx.x; f < F; f += blockDim.x)
        atomicAdd(&od[f], hs[f] * c);
}

// act: 0 none, 1 relu, 2 elu
__global__ void bias_act(float* X, const float* __restrict__ b, int n, int F, int act) {
    size_t idx = (size_t)blockIdx.x * blockDim.x + threadIdx.x;
    if (idx >= (size_t)n * F) return;
    int f = (int)(idx % F);
    float v = X[idx] + (b ? b[f] : 0.f);
    if (act == 1)      v = v > 0.f ? v : 0.f;
    else if (act == 2) v = v > 0.f ? v : (expf(v) - 1.f);
    X[idx] = v;
}

// ---------------------------------------------------------------------------
// BatchNorm (two pass)
// ---------------------------------------------------------------------------
__global__ void zero_f32(float* p, size_t n) {
    size_t i = (size_t)blockIdx.x * blockDim.x + threadIdx.x;
    if (i < n) p[i] = 0.f;
}
__global__ void neginf_f32(float* p, size_t n) {
    size_t i = (size_t)blockIdx.x * blockDim.x + threadIdx.x;
    if (i < n) p[i] = -__builtin_inff();
}
#define BN_RPB 64
__global__ void bn_partial(const float* __restrict__ X, float* sum, float* sumsq,
                           int n, int F) {
    int r0 = blockIdx.x * BN_RPB;
    int rend = r0 + BN_RPB; if (rend > n) rend = n;
    for (int f = threadIdx.x; f < F; f += blockDim.x) {
        float s = 0.f, q = 0.f;
        for (int r = r0; r < rend; ++r) {
            float v = X[(size_t)r * F + f];
            s += v; q += v * v;
        }
        atomicAdd(&sum[f], s);
        atomicAdd(&sumsq[f], q);
    }
}
__global__ void bn_apply(float* X, const float* __restrict__ sum,
                         const float* __restrict__ sumsq,
                         const float* __restrict__ g, const float* __restrict__ b,
                         int n, int F) {
    size_t idx = (size_t)blockIdx.x * blockDim.x + threadIdx.x;
    if (idx >= (size_t)n * F) return;
    int f = (int)(idx % F);
    float mu  = sum[f] / n;
    float var = sumsq[f] / n - mu * mu;
    X[idx] = (X[idx] - mu) * rsqrtf(var + BN_EPS) * g[f] + b[f];
}

// ---------------------------------------------------------------------------
// GAT (H = 1<<hshift heads, F = 1<<fshift feats/head)
// ---------------------------------------------------------------------------
__global__ void gat_alpha(const float* __restrict__ hfeat, const float* __restrict__ a_s,
                          const float* __restrict__ a_d, float* al_s, float* al_d,
                          int n, int H, int F) {
    int wid  = (blockIdx.x * blockDim.x + threadIdx.x) >> 5;
    int lane = threadIdx.x & 31;
    if (wid >= n * H) return;
    int node = wid / H, h = wid - node * H;
    const float* hp = hfeat + (size_t)node * H * F + (size_t)h * F;
    const float* as = a_s + (size_t)h * F;
    const float* ad = a_d + (size_t)h * F;
    float ss = 0.f, sd = 0.f;
    for (int f = lane; f < F; f += 32) {
        float v = hp[f];
        ss += v * as[f];
        sd += v * ad[f];
    }
#pragma unroll
    for (int o = 16; o > 0; o >>= 1) {
        ss += __shfl_xor(ss, o, 32);
        sd += __shfl_xor(sd, o, 32);
    }
    if (lane == 0) { al_s[wid] = ss; al_d[wid] = sd; }
}

__global__ void gat_logits(const long long* __restrict__ ei, const float* __restrict__ al_s,
                           const float* __restrict__ al_d, float* elog, float* m,
                           int E, int n, int hshift) {
    const int H = 1 << hshift;
    size_t idx = (size_t)blockIdx.x * blockDim.x + threadIdx.x;
    size_t tot = (size_t)(E + n) << hshift;
    if (idx >= tot) return;
    int e = (int)(idx >> hshift), h = (int)(idx & (H - 1));
    long long s, d;
    if (e < E) { s = ei[e]; d = ei[(size_t)E + e]; } else { s = d = e - E; }
    float v = al_s[((size_t)s << hshift) + h] + al_d[((size_t)d << hshift) + h];
    v = v > 0.f ? v : NEG_SLOPE * v;
    elog[idx] = v;
    atomicMaxF(&m[((size_t)d << hshift) + h], v);
}

__global__ void gat_exp(const long long* __restrict__ ei, float* elog,
                        const float* __restrict__ m, float* denom,
                        int E, int n, int hshift) {
    const int H = 1 << hshift;
    size_t idx = (size_t)blockIdx.x * blockDim.x + threadIdx.x;
    size_t tot = (size_t)(E + n) << hshift;
    if (idx >= tot) return;
    int e = (int)(idx >> hshift), h = (int)(idx & (H - 1));
    long long d = (e < E) ? ei[(size_t)E + e] : (long long)(e - E);
    float w = expf(elog[idx] - m[((size_t)d << hshift) + h]);
    elog[idx] = w;
    atomicAdd(&denom[((size_t)d << hshift) + h], w);
}

__global__ void gat_agg(const long long* __restrict__ ei, const float* __restrict__ elog,
                        const float* __restrict__ denom, const float* __restrict__ hfeat,
                        float* __restrict__ out, int E, int n, int hshift, int fshift) {
    int e = blockIdx.x;                               // 0 .. E+n-1
    long long s, d;
    if (e < E) { s = ei[e]; d = ei[(size_t)E + e]; } else { s = d = e - E; }
    const int HF = 1 << (hshift + fshift);
    const float* hs = hfeat + ((size_t)s << (hshift + fshift));
    float* od = out + ((size_t)d << (hshift + fshift));
    for (int f = threadIdx.x; f < HF; f += blockDim.x) {
        int h = f >> fshift;
        float alpha = elog[((size_t)e << hshift) + h] /
                      denom[((size_t)d << hshift) + h];
        atomicAdd(&od[f], alpha * hs[f]);
    }
}

// ---------------------------------------------------------------------------
// misc
// ---------------------------------------------------------------------------
__global__ void add_inplace(float* a, const float* __restrict__ b, size_t n) {
    size_t i = (size_t)blockIdx.x * blockDim.x + threadIdx.x;
    if (i < n) a[i] += b[i];
}
__global__ void log_softmax_rows(const float* __restrict__ X, float* __restrict__ out,
                                 int n, int ld, int C) {
    int r = blockIdx.x * blockDim.x + threadIdx.x;
    if (r >= n) return;
    const float* x = X + (size_t)r * ld;
    float mx = -__builtin_inff();
    for (int c = 0; c < C; ++c) mx = fmaxf(mx, x[c]);
    float s = 0.f;
    for (int c = 0; c < C; ++c) s += expf(x[c] - mx);
    float ls = logf(s);
    for (int c = 0; c < C; ++c) out[(size_t)r * C + c] = x[c] - mx - ls;
}

// ---------------------------------------------------------------------------
// host orchestration
// ---------------------------------------------------------------------------
static inline dim3 ew_grid(size_t cnt) { return dim3((unsigned)((cnt + 255) / 256)); }

extern "C" void kernel_launch(void* const* d_in, const int* in_sizes, int n_in,
                              void* d_out, int out_size, void* d_ws, size_t ws_size,
                              hipStream_t stream) {
    (void)n_in; (void)out_size; (void)ws_size;
    const float*     x       = (const float*)d_in[0];
    const long long* ei      = (const long long*)d_in[1];
    const float*     gcn1_w  = (const float*)d_in[2];
    const float*     gcn1_b  = (const float*)d_in[3];
    const float*     gcn2_w  = (const float*)d_in[4];
    const float*     gcn2_b  = (const float*)d_in[5];
    const float*     bng_g   = (const float*)d_in[6];
    const float*     bng_b   = (const float*)d_in[7];
    const float*     proj_w  = (const float*)d_in[8];
    const float*     proj_b  = (const float*)d_in[9];
    const float*     gat1_w  = (const float*)d_in[10];
    const float*     gat1_as = (const float*)d_in[11];
    const float*     gat1_ad = (const float*)d_in[12];
    const float*     gat1_b  = (const float*)d_in[13];
    const float*     gat2_w  = (const float*)d_in[14];
    const float*     gat2_as = (const float*)d_in[15];
    const float*     gat2_ad = (const float*)d_in[16];
    const float*     gat2_b  = (const float*)d_in[17];
    const float*     bnt_g   = (const float*)d_in[18];
    const float*     bnt_b   = (const float*)d_in[19];
    const float*     fc1_w   = (const float*)d_in[20];
    const float*     fc1_b   = (const float*)d_in[21];
    const float*     fc2_w   = (const float*)d_in[22];
    const float*     fc2_b   = (const float*)d_in[23];

    const int D = 512, H = 256, HEADS = 4, HF = 1024, OUTC = 10, OUTP = 16;
    const int N  = in_sizes[0] / D;     // 50000 (multiple of 16)
    const int E  = in_sizes[1] / 2;
    const int EN = E + N;

    // --- workspace carve-out (256B aligned) ---
    char* ws = (char*)d_ws;
    size_t off = 0;
    auto alloc = [&](size_t bytes) -> char* {
        off = (off + 255) & ~(size_t)255;
        char* p = ws + off; off += bytes; return p;
    };
    unsigned short* pk_gcn1 = (unsigned short*)alloc((size_t)H    * D   * 2);
    unsigned short* pk_gcn2 = (unsigned short*)alloc((size_t)2*H  * H   * 2);
    unsigned short* pk_proj = (unsigned short*)alloc((size_t)H    * 2*H * 2);
    unsigned short* pk_gat1 = (unsigned short*)alloc((size_t)HF   * H   * 2);
    unsigned short* pk_gat2 = (unsigned short*)alloc((size_t)H    * HF  * 2);
    unsigned short* pk_fc1  = (unsigned short*)alloc((size_t)128  * H   * 2);
    unsigned short* pk_fc2  = (unsigned short*)alloc((size_t)OUTP * 128 * 2);
    unsigned short* Abf = (unsigned short*)alloc((size_t)N * HF * 2);  // bf16 A staging
    float* dinv = (float*)alloc((size_t)N * 4);
    float* R0   = (float*)alloc((size_t)N * HF * 4);   // t1 -> hfeat1 -> hfeat2 -> fc2out
    float* R1   = (float*)alloc((size_t)N * HF * 4);   // t0 -> h2 -> g1 -> f1
    float* R2   = (float*)alloc((size_t)N * H  * 4);   // h1 -> x_proj -> xc
    float* R3   = (float*)alloc((size_t)N * H  * 4);   // g2
    float* al1s = (float*)alloc((size_t)N * HEADS * 4);
    float* al1d = (float*)alloc((size_t)N * HEADS * 4);
    float* m1   = (float*)alloc((size_t)N * HEADS * 4);
    float* den1 = (float*)alloc((size_t)N * HEADS * 4);
    float* al2s = (float*)alloc((size_t)N * 4);
    float* al2d = (float*)alloc((size_t)N * 4);
    float* m2   = (float*)alloc((size_t)N * 4);
    float* den2 = (float*)alloc((size_t)N * 4);
    float* elog = (float*)alloc((size_t)EN * HEADS * 4);
    float* bns  = (float*)alloc(512 * 4);
    float* bnq  = (float*)alloc(512 * 4);

    auto pack = [&](const float* W, unsigned short* Bt, int K, int Nact, int Npad) {
        int tot = Npad * K;
        pack_weight_bf16<<<ew_grid(tot), 256, 0, stream>>>(W, Bt, K, Nact, Npad);
    };
    // convert A f32->bf16, then GEMM from bf16
    auto gemm = [&](const float* A, const unsigned short* Bt, const float* bias,
                    float* C, int M, int K, int Npad, int Nact, int act) {
        size_t npairs = ((size_t)M * K) / 2;
        f32_to_bf16<<<ew_grid(npairs), 256, 0, stream>>>(A, Abf, npairs);
        dim3 grid((Npad / 16 + 7) / 8, M / 16);
        wmma_gemm_bf16<<<grid, 256, 0, stream>>>(Abf, Bt, bias, C, M, K, Npad, Nact, act);
    };

    // weight packs
    pack(gcn1_w, pk_gcn1, D,   H,    H);
    pack(gcn2_w, pk_gcn2, H,   2*H,  2*H);
    pack(proj_w, pk_proj, 2*H, H,    H);
    pack(gat1_w, pk_gat1, H,   HF,   HF);
    pack(gat2_w, pk_gat2, HF,  H,    H);
    pack(fc1_w,  pk_fc1,  H,   128,  128);
    pack(fc2_w,  pk_fc2,  128, OUTC, OUTP);

    // degree -> dinv
    deg_init<<<ew_grid(N), 256, 0, stream>>>(dinv, N);
    deg_accum<<<ew_grid(E), 256, 0, stream>>>(ei + E, dinv, E);
    deg_finalize<<<ew_grid(N), 256, 0, stream>>>(dinv, N);

    // ---- GCN layer 1: t0 = x@W (R1) ; h1 = relu(agg+b) (R2)
    gemm(x, pk_gcn1, nullptr, R1, N, D, H, H, 0);
    gcn_self<<<ew_grid((size_t)N * H), 256, 0, stream>>>(R1, dinv, R2, N, H);
    gcn_edge_agg<<<E, 256, 0, stream>>>(R1, ei, dinv, R2, E, H);
    bias_act<<<ew_grid((size_t)N * H), 256, 0, stream>>>(R2, gcn1_b, N, H, 1);

    // ---- GCN layer 2: t1 = h1@W (R0) ; h2 = relu(agg+b) (R1)
    gemm(R2, pk_gcn2, nullptr, R0, N, H, 2*H, 2*H, 0);
    gcn_self<<<ew_grid((size_t)N * 2*H), 256, 0, stream>>>(R0, dinv, R1, N, 2*H);
    gcn_edge_agg<<<E, 256, 0, stream>>>(R0, ei, dinv, R1, E, 2*H);
    bias_act<<<ew_grid((size_t)N * 2*H), 256, 0, stream>>>(R1, gcn2_b, N, 2*H, 1);

    // ---- BN(gcn) in place on R1
    zero_f32<<<ew_grid(2*H), 256, 0, stream>>>(bns, 2*H);
    zero_f32<<<ew_grid(2*H), 256, 0, stream>>>(bnq, 2*H);
    bn_partial<<<(N + BN_RPB - 1) / BN_RPB, 256, 0, stream>>>(R1, bns, bnq, N, 2*H);
    bn_apply<<<ew_grid((size_t)N * 2*H), 256, 0, stream>>>(R1, bns, bnq, bng_g, bng_b, N, 2*H);

    // ---- x_proj = h@proj_w + b (R2)
    gemm(R1, pk_proj, proj_b, R2, N, 2*H, H, H, 0);

    // ---- GAT layer 1 (heads=4, F=256): hfeat1 (R0), out g1 (R1)
    gemm(R2, pk_gat1, nullptr, R0, N, H, HF, HF, 0);
    gat_alpha<<<ew_grid((size_t)N * HEADS * 32), 256, 0, stream>>>(R0, gat1_as, gat1_ad,
                                                                   al1s, al1d, N, HEADS, H);
    neginf_f32<<<ew_grid((size_t)N * HEADS), 256, 0, stream>>>(m1, (size_t)N * HEADS);
    zero_f32<<<ew_grid((size_t)N * HEADS), 256, 0, stream>>>(den1, (size_t)N * HEADS);
    zero_f32<<<ew_grid((size_t)N * HF), 256, 0, stream>>>(R1, (size_t)N * HF);
    gat_logits<<<ew_grid((size_t)EN * HEADS), 256, 0, stream>>>(ei, al1s, al1d, elog, m1, E, N, 2);
    gat_exp<<<ew_grid((size_t)EN * HEADS), 256, 0, stream>>>(ei, elog, m1, den1, E, N, 2);
    gat_agg<<<EN, 256, 0, stream>>>(ei, elog, den1, R0, R1, E, N, 2, 8);
    bias_act<<<ew_grid((size_t)N * HF), 256, 0, stream>>>(R1, gat1_b, N, HF, 2);  // +b, ELU

    // ---- GAT layer 2 (heads=1, F=256): hfeat2 (R0), out g2 (R3)
    gemm(R1, pk_gat2, nullptr, R0, N, HF, H, H, 0);
    gat_alpha<<<ew_grid((size_t)N * 32), 256, 0, stream>>>(R0, gat2_as, gat2_ad,
                                                           al2s, al2d, N, 1, H);
    neginf_f32<<<ew_grid(N), 256, 0, stream>>>(m2, (size_t)N);
    zero_f32<<<ew_grid(N), 256, 0, stream>>>(den2, (size_t)N);
    zero_f32<<<ew_grid((size_t)N * H), 256, 0, stream>>>(R3, (size_t)N * H);
    gat_logits<<<ew_grid((size_t)EN), 256, 0, stream>>>(ei, al2s, al2d, elog, m2, E, N, 0);
    gat_exp<<<ew_grid((size_t)EN), 256, 0, stream>>>(ei, elog, m2, den2, E, N, 0);
    gat_agg<<<EN, 256, 0, stream>>>(ei, elog, den2, R0, R3, E, N, 0, 8);
    bias_act<<<ew_grid((size_t)N * H), 256, 0, stream>>>(R3, gat2_b, N, H, 2);    // +b, ELU

    // ---- BN(gat) in place on R3, then xc = x_proj + g (R2)
    zero_f32<<<ew_grid(H), 256, 0, stream>>>(bns, H);
    zero_f32<<<ew_grid(H), 256, 0, stream>>>(bnq, H);
    bn_partial<<<(N + BN_RPB - 1) / BN_RPB, 256, 0, stream>>>(R3, bns, bnq, N, H);
    bn_apply<<<ew_grid((size_t)N * H), 256, 0, stream>>>(R3, bns, bnq, bnt_g, bnt_b, N, H);
    add_inplace<<<ew_grid((size_t)N * H), 256, 0, stream>>>(R2, R3, (size_t)N * H);

    // ---- MLP head
    gemm(R2, pk_fc1, fc1_b, R1, N, H, 128, 128, 1);        // relu fused
    gemm(R1, pk_fc2, fc2_b, R0, N, 128, OUTP, OUTC, 0);    // padded N=16, ld=16
    log_softmax_rows<<<ew_grid(N), 256, 0, stream>>>(R0, (float*)d_out, N, OUTP, OUTC);
}
